// AttentionHead_7894149890101
// MI455X (gfx1250) — matrix operand order repaired and checked by reading, not compiled, not run on hardware
//
#include <hip/hip_runtime.h>
#include <hip/hip_bf16.h>

#define BB 16
#define SS 1024
#define HH 1024
#define DD 128

typedef __attribute__((ext_vector_type(16))) _Float16 v16h;
typedef __attribute__((ext_vector_type(8)))  _Float16 v8h;
typedef __attribute__((ext_vector_type(8)))  float    v8f;
typedef __attribute__((ext_vector_type(4)))  float    v4f;
typedef __attribute__((ext_vector_type(4)))  unsigned int u32x4;
typedef __attribute__((ext_vector_type(8)))  int      i32x8;
typedef __attribute__((ext_vector_type(4)))  int      i32x4;

// D = A(16x32 f16) * B(32x16 f16) + C(16x16 f32)
__device__ __forceinline__ v8f wmma_f16(v16h a, v16h b, v8f c) {
  return __builtin_amdgcn_wmma_f32_16x16x32_f16(
      /*neg_a=*/false, a, /*neg_b=*/false, b,
      /*c_mod=*/(short)0, c, /*reuse_a=*/false, /*reuse_b=*/false);
}

__device__ __forceinline__ v16h pack16(v8h lo, v8h hi) {
  v16h r;
#pragma unroll
  for (int i = 0; i < 8; ++i) { r[i] = lo[i]; r[8 + i] = hi[i]; }
  return r;
}

// TDM: DMA a dense 2-D tile [nrows x 128 f32] (row stride 128 elems) into LDS.
// D# per CDNA5 ISA ch.8: group0 = count/lds_addr/global_addr/type,
// group1 = data_size/dims/tile/strides, trailing groups zero (2-D tensor).
// This toolchain's builtin takes 6 args (g0, g1, g2, g3, extra, cpol).
__device__ __forceinline__ void tdm_load_rows_f32(const float* g, unsigned lds_off,
                                                  int nrows) {
  unsigned long long ga = (unsigned long long)(size_t)g;
  u32x4 g0;
  g0[0] = 1u;                                           // count=1, user mode
  g0[1] = lds_off;                                      // lds_addr (bytes)
  g0[2] = (unsigned)ga;                                 // global_addr[31:0]
  g0[3] = (unsigned)((ga >> 32) & 0x1FFFFFFu) | (2u << 30);  // addr[56:32], type=2
  i32x8 g1;
  g1[0] = (int)(2u << 16);                // data_size = 4 bytes
  g1[1] = (int)(128u << 16);              // tensor_dim0[15:0] = 128 (d)
  g1[2] = (int)((unsigned)nrows << 16);   // tensor_dim1[15:0] = nrows (t)
  g1[3] = (int)(128u << 16);              // tile_dim0 = 128
  g1[4] = nrows;                          // tile_dim1 = nrows
  g1[5] = 128;                            // tensor_dim0_stride = 128 elems
  g1[6] = 0;
  g1[7] = 0;
  i32x4 z4 = {0, 0, 0, 0};
  i32x8 z8 = {0, 0, 0, 0, 0, 0, 0, 0};
  __builtin_amdgcn_tensor_load_to_lds(g0, g1, z4, z4, z8, 0);
}

// ---------------------------------------------------------------------------
// Kernel 1: projection  Y[r,d] = sum_h X[r,h]*W[d,h] + bias[d]   (f16 output)
// Block: 16 token rows x 128 d-cols (8 waves, one 16-wide d-tile per wave).
// ---------------------------------------------------------------------------
__global__ __launch_bounds__(256) void proj_kernel(
    const float* __restrict__ X, const float* __restrict__ W,
    const float* __restrict__ bias, _Float16* __restrict__ Y)
{
  const int lane = threadIdx.x & 31;
  const int wv   = threadIdx.x >> 5;
  const int r0   = blockIdx.x * 16;
  const int d0   = wv * 16;
  const int n    = lane & 15;
  const int ab0  = (lane < 16) ? 0 : 8;   // A-matrix interleave base
  const int kh   = (lane < 16) ? 0 : 16;  // unused; B uses ab0 order like A
  const float* xrow = X + (size_t)(r0 + (lane & 15)) * HH;
  const float* wrow = W + (size_t)(d0 + n) * HH;

  v8f acc = {};
  for (int h0 = 0; h0 < HH; h0 += 32) {
    v4f xa0 = *(const v4f*)(xrow + h0 + ab0);
    v4f xa1 = *(const v4f*)(xrow + h0 + ab0 + 4);
    v4f xb0 = *(const v4f*)(xrow + h0 + 16 + ab0);
    v4f xb1 = *(const v4f*)(xrow + h0 + 16 + ab0 + 4);
    v16h a;
#pragma unroll
    for (int i = 0; i < 4; ++i) {
      a[i]      = (_Float16)xa0[i]; a[4 + i]  = (_Float16)xa1[i];
      a[8 + i]  = (_Float16)xb0[i]; a[12 + i] = (_Float16)xb1[i];
    }
    const v4f* wp = (const v4f*)(wrow + h0 + ab0);
    v4f w0 = wp[0], w1 = wp[1], w2 = wp[4], w3 = wp[5];
    v16h b;
#pragma unroll
    for (int i = 0; i < 4; ++i) {
      b[i]      = (_Float16)w0[i]; b[4 + i]  = (_Float16)w1[i];
      b[8 + i]  = (_Float16)w2[i]; b[12 + i] = (_Float16)w3[i];
    }
    acc = wmma_f16(a, b, acc);
  }
  const float bv = bias[d0 + n];
  const int mb = (lane < 16) ? 0 : 8;
#pragma unroll
  for (int j = 0; j < 8; ++j)
    Y[(size_t)(r0 + mb + j) * DD + d0 + n] = (_Float16)(acc[j] + bv);
}

// ---------------------------------------------------------------------------
// Kernel 2: attn1[b,s,t] = sum_d q[b,s,d]*k[b,t,d]  -> f32 scores buffer
// ---------------------------------------------------------------------------
__global__ __launch_bounds__(256) void attn1_kernel(
    const _Float16* __restrict__ Q, const _Float16* __restrict__ K,
    float* __restrict__ S1)
{
  const int lane = threadIdx.x & 31;
  const int wv   = threadIdx.x >> 5;
  const int s0   = blockIdx.x * 16;
  const int t0   = blockIdx.y * 128 + wv * 16;
  const int b    = blockIdx.z;
  const int n = lane & 15, ab0 = (lane < 16) ? 0 : 8, kh = (lane < 16) ? 0 : 16;
  const _Float16* qrow = Q + ((size_t)b * SS + s0 + (lane & 15)) * DD;
  const _Float16* kcol = K + ((size_t)b * SS + t0 + n) * DD;

  v8f acc = {};
#pragma unroll
  for (int d0 = 0; d0 < DD; d0 += 32) {
    v16h a = pack16(*(const v8h*)(qrow + d0 + ab0),
                    *(const v8h*)(qrow + d0 + 16 + ab0));
    v16h bm = pack16(*(const v8h*)(kcol + d0 + ab0),
                     *(const v8h*)(kcol + d0 + 16 + ab0));
    acc = wmma_f16(a, bm, acc);
  }
  const int mb = (lane < 16) ? 0 : 8;
#pragma unroll
  for (int j = 0; j < 8; ++j)
    S1[((size_t)b * SS + s0 + mb + j) * SS + t0 + n] = acc[j];
}

// ---------------------------------------------------------------------------
// Kernel 3: one block per s.  attn2 = q[:,s,:] (16 batch rows) x k_bias[s]^T,
// combine with attn1, scale, mask, softmax over t, store w as f16.
// ---------------------------------------------------------------------------
__global__ __launch_bounds__(256) void attn2_softmax_kernel(
    const _Float16* __restrict__ Q, const float* __restrict__ KB,
    const float* __restrict__ S1, const int* __restrict__ mask,
    _Float16* __restrict__ Wout)
{
  __shared__ float sc[16 * SS];  // 64 KB row-block [b=16][t=1024]
  const int s    = blockIdx.x;
  const int lane = threadIdx.x & 31;
  const int wv   = threadIdx.x >> 5;
  const int n = lane & 15, ab0 = (lane < 16) ? 0 : 8, kh = (lane < 16) ? 0 : 16;
  const int mb = (lane < 16) ? 0 : 8;
  const float scale = 0.08838834764831845f;  // 1/sqrt(128)

  const _Float16* qrow = Q + ((size_t)(lane & 15) * SS + s) * DD;
  v16h atile[4];
#pragma unroll
  for (int dt = 0; dt < 4; ++dt)
    atile[dt] = pack16(*(const v8h*)(qrow + dt * 32 + ab0),
                       *(const v8h*)(qrow + dt * 32 + 16 + ab0));

  for (int tc = 0; tc < SS / 128; ++tc) {
    const int t0 = tc * 128 + wv * 16;
    const float* kbcol = KB + ((size_t)s * SS + t0 + n) * DD;  // streamed once
    v8f acc = {};
#pragma unroll
    for (int dt = 0; dt < 4; ++dt) {
      const v4f* kp = (const v4f*)(kbcol + dt * 32 + ab0);
      v4f w0 = __builtin_nontemporal_load(kp + 0);
      v4f w1 = __builtin_nontemporal_load(kp + 1);
      v4f w2 = __builtin_nontemporal_load(kp + 4);
      v4f w3 = __builtin_nontemporal_load(kp + 5);
      v16h bm;
#pragma unroll
      for (int i = 0; i < 4; ++i) {
        bm[i]      = (_Float16)w0[i]; bm[4 + i]  = (_Float16)w1[i];
        bm[8 + i]  = (_Float16)w2[i]; bm[12 + i] = (_Float16)w3[i];
      }
      acc = wmma_f16(atile[dt], bm, acc);
    }
#pragma unroll
    for (int j = 0; j < 8; ++j) {
      const int m = mb + j;  // batch index
      float v = (acc[j] + S1[((size_t)m * SS + s) * SS + t0 + n]) * scale;
      if (mask[m * SS + t0 + n] == 0) v = -1e9f;
      sc[m * SS + t0 + n] = v;
    }
  }
  __syncthreads();

  const int r = threadIdx.x >> 4;
  const int c = threadIdx.x & 15;
  float mx = -1e30f;
  for (int t = c; t < SS; t += 16) mx = fmaxf(mx, sc[r * SS + t]);
#pragma unroll
  for (int off = 8; off; off >>= 1) mx = fmaxf(mx, __shfl_xor(mx, off, 16));
  float sum = 0.f;
  for (int t = c; t < SS; t += 16) {
    float e = __expf(sc[r * SS + t] - mx);
    sc[r * SS + t] = e;
    sum += e;
  }
#pragma unroll
  for (int off = 8; off; off >>= 1) sum += __shfl_xor(sum, off, 16);
  const float inv = 1.0f / sum;
  for (int t = c; t < SS; t += 16)
    Wout[((size_t)r * SS + s) * SS + t] = (_Float16)(sc[r * SS + t] * inv);
}

// ---------------------------------------------------------------------------
// Kernel 4: values1[b,s,d] = sum_t w[b,s,t]*v[b,t,d]  (writes out)
// ---------------------------------------------------------------------------
__global__ __launch_bounds__(256) void values1_kernel(
    const _Float16* __restrict__ Wt, const _Float16* __restrict__ V,
    float* __restrict__ Out)
{
  const int lane = threadIdx.x & 31;
  const int wv   = threadIdx.x >> 5;
  const int s0   = blockIdx.x * 16;
  const int b    = blockIdx.y;
  const int d0   = wv * 16;
  const int n = lane & 15, ab0 = (lane < 16) ? 0 : 8, kh = (lane < 16) ? 0 : 16;
  const _Float16* wrow = Wt + ((size_t)b * SS + s0 + (lane & 15)) * SS;

  v8f acc = {};
  for (int t0 = 0; t0 < SS; t0 += 32) {
    v16h a = pack16(*(const v8h*)(wrow + t0 + ab0),
                    *(const v8h*)(wrow + t0 + 16 + ab0));
    v16h bm;  // B[k=t][n=d] = V[b,t,d0+n] (V is L2-resident, 4 MB)
#pragma unroll
    for (int i = 0; i < 8; ++i) {
      const int t = t0 + ab0 + 2 * i + (i >= 4 ? 8 : 0);
      bm[2 * i]     = V[((size_t)b * SS + t) * DD + d0 + n];
      bm[2 * i + 1] = V[((size_t)b * SS + t + 1) * DD + d0 + n];
    }
    acc = wmma_f16(a, bm, acc);
  }
  const int mb = (lane < 16) ? 0 : 8;
#pragma unroll
  for (int j = 0; j < 8; ++j)
    Out[((size_t)b * SS + s0 + mb + j) * DD + d0 + n] = acc[j];
}

// ---------------------------------------------------------------------------
// Kernel 5: one block per s.  values2 = w[:,s,:] (16xB) x v_bias[s]; Out +=.
// v_bias[s] (512 KB) is streamed through LDS with a double-buffered TDM
// pipeline: wave 0 issues TENSOR_LOAD_TO_LDS for chunk i+1, waits
// tensorcnt<=1 (in-order => chunk i landed), then a barrier publishes it.
// ---------------------------------------------------------------------------
__global__ __launch_bounds__(256) void values2_kernel(
    const _Float16* __restrict__ Wt, const float* __restrict__ VB,
    float* __restrict__ Out)
{
  __shared__ float vb[2][32 * DD];  // 2 x 16 KB tiles [32 t][128 d]
  const int s    = blockIdx.x;
  const int lane = threadIdx.x & 31;
  const int wv   = threadIdx.x >> 5;
  const int d0   = wv * 16;
  const int n = lane & 15, ab0 = (lane < 16) ? 0 : 8, kh = (lane < 16) ? 0 : 16;
  const _Float16* wrow = Wt + ((size_t)(lane & 15) * SS + s) * SS;  // row = batch
  const float* vbase = VB + (size_t)s * SS * DD;
  const unsigned lds0 = (unsigned)(size_t)&vb[0][0];
  const unsigned lds1 = (unsigned)(size_t)&vb[1][0];

  if (wv == 0) tdm_load_rows_f32(vbase, lds0, 32);  // prologue: chunk 0

  v8f acc = {};
  for (int it = 0; it < SS / 32; ++it) {
    if (wv == 0) {
      if (it + 1 < SS / 32) {
        tdm_load_rows_f32(vbase + (size_t)(it + 1) * 32 * DD,
                          ((it + 1) & 1) ? lds1 : lds0, 32);
        __builtin_amdgcn_s_wait_tensorcnt(1);  // chunk `it` complete
      } else {
        __builtin_amdgcn_s_wait_tensorcnt(0);
      }
    }
    __syncthreads();  // publish chunk `it` to all waves

    const int t0 = it * 32;
    v16h a = pack16(*(const v8h*)(wrow + t0 + ab0),
                    *(const v8h*)(wrow + t0 + 16 + ab0));
    const float* buf = vb[it & 1];
    v16h bm;  // B[k=t][n=d] from LDS tile
#pragma unroll
    for (int i = 0; i < 8; ++i) {
      const int tt = ab0 + 2 * i + (i >= 4 ? 8 : 0);
      bm[2 * i]     = (_Float16)buf[tt * DD + d0 + n];
      bm[2 * i + 1] = (_Float16)buf[(tt + 1) * DD + d0 + n];
    }
    acc = wmma_f16(a, bm, acc);
    __syncthreads();  // all reads done before TDM may overwrite this buffer
  }
  const int mb = (lane < 16) ? 0 : 8;
#pragma unroll
  for (int j = 0; j < 8; ++j) {
    const size_t idx = ((size_t)(mb + j) * SS + s) * DD + d0 + n;
    Out[idx] += acc[j];  // race-free: this block owns Out[:,s,d-tile]
  }
}

// ---------------------------------------------------------------------------
extern "C" void kernel_launch(void* const* d_in, const int* in_sizes, int n_in,
                              void* d_out, int out_size, void* d_ws, size_t ws_size,
                              hipStream_t stream) {
  (void)in_sizes; (void)n_in; (void)out_size; (void)ws_size;
  const float* query  = (const float*)d_in[0];
  const float* key    = (const float*)d_in[1];
  const float* value  = (const float*)d_in[2];
  const int*   mask   = (const int*)d_in[3];
  const float* Wq = (const float*)d_in[4];
  const float* bq = (const float*)d_in[5];
  const float* Wk = (const float*)d_in[6];
  const float* bk = (const float*)d_in[7];
  const float* Wv = (const float*)d_in[8];
  const float* bv = (const float*)d_in[9];
  const float* k_bias = (const float*)d_in[10];
  const float* v_bias = (const float*)d_in[11];
  float* out = (float*)d_out;

  char* ws = (char*)d_ws;
  _Float16* qf  = (_Float16*)(ws);                        //  4 MB  [B,S,D] f16
  _Float16* kf  = (_Float16*)(ws + ((size_t)4  << 20));   //  4 MB
  _Float16* vf  = (_Float16*)(ws + ((size_t)8  << 20));   //  4 MB
  float*    s1  = (float*)   (ws + ((size_t)12 << 20));   // 64 MB  [B,S,S] f32
  _Float16* wsm = (_Float16*)(ws + ((size_t)76 << 20));   // 32 MB  [B,S,S] f16

  const dim3 blk(256);
  proj_kernel<<<dim3(BB * SS / 16), blk, 0, stream>>>(query, Wq, bq, qf);
  proj_kernel<<<dim3(BB * SS / 16), blk, 0, stream>>>(key,   Wk, bk, kf);
  proj_kernel<<<dim3(BB * SS / 16), blk, 0, stream>>>(value, Wv, bv, vf);
  attn1_kernel<<<dim3(SS / 16, SS / 128, BB), blk, 0, stream>>>(qf, kf, s1);
  attn2_softmax_kernel<<<dim3(SS), blk, 0, stream>>>(qf, k_bias, s1, mask, wsm);
  values1_kernel<<<dim3(SS / 16, BB), blk, 0, stream>>>(wsm, vf, out);
  values2_kernel<<<dim3(SS), blk, 0, stream>>>(wsm, v_bias, out);
}
